// vAutoencoder_27642409517588
// MI455X (gfx1250) — matrix operand order repaired and checked by reading, not compiled, not run on hardware
//
#include <hip/hip_runtime.h>
#include <hip/hip_bf16.h>

typedef __attribute__((ext_vector_type(16))) __bf16 v16bf_t;
typedef __attribute__((ext_vector_type(8)))  __bf16 v8bf_t;
typedef __attribute__((ext_vector_type(4)))  __bf16 v4bf_t;
typedef __attribute__((ext_vector_type(2)))  __bf16 v2bf_t;
typedef __attribute__((ext_vector_type(8)))  float  v8f_t;

union AFragU { v16bf_t v; v8bf_t h[2]; };

#define BM 128
#define BN 256
#define BK 32
#define LDP 40   // LDS row pitch in bf16 elements (32 + 8 pad, 80B rows)

struct StageRegs {
  float4 a[4];    // A tile: 4 x float4 per thread
  float4 b0[4];   // B tile row k even
  float4 b1[4];   // B tile row k odd
};

__device__ __forceinline__ void stage_load_global(
    const float* __restrict__ A, const float* __restrict__ B,
    int K, int N, int bm0, int bn0, int k0, int tid, StageRegs& r) {
#pragma unroll
  for (int i = 0; i < 4; ++i) {   // A: 128x32 f32, 1024 float4
    int idx = tid + i * 256;
    int row = idx >> 3;           // 0..127
    int c4  = (idx & 7) << 2;     // 0..28
    r.a[i] = *(const float4*)(A + (size_t)(bm0 + row) * K + k0 + c4);
  }
#pragma unroll
  for (int i = 0; i < 4; ++i) {   // B: 32x256 f32 as 16 row-pairs x 64 col-groups
    int idx = tid + i * 256;
    int kp  = idx >> 6;           // 0..15 (row pair)
    int c4  = (idx & 63) << 2;    // 0..252
    const float* base = B + (size_t)(k0 + 2 * kp) * N + bn0 + c4;
    r.b0[i] = *(const float4*)(base);
    r.b1[i] = *(const float4*)(base + N);
  }
}

__device__ __forceinline__ void stage_store_lds(
    __bf16* __restrict__ As, __bf16* __restrict__ Bs, int tid,
    const StageRegs& r) {
#pragma unroll
  for (int i = 0; i < 4; ++i) {
    int idx = tid + i * 256;
    int row = idx >> 3;
    int c4  = (idx & 7) << 2;
    const float* f = (const float*)&r.a[i];
    v4bf_t hv;
    hv[0] = (__bf16)f[0]; hv[1] = (__bf16)f[1];
    hv[2] = (__bf16)f[2]; hv[3] = (__bf16)f[3];
    *(v4bf_t*)(&As[row * LDP + c4]) = hv;
  }
#pragma unroll
  for (int i = 0; i < 4; ++i) {
    int idx = tid + i * 256;
    int kp  = idx >> 6;
    int c4  = (idx & 63) << 2;
    const float* f0 = (const float*)&r.b0[i];
    const float* f1 = (const float*)&r.b1[i];
#pragma unroll
    for (int j = 0; j < 4; ++j) {   // transposed: Bs[n][k], pack (k, k+1)
      v2bf_t p;
      p[0] = (__bf16)f0[j];
      p[1] = (__bf16)f1[j];
      *(v2bf_t*)(&Bs[(c4 + j) * LDP + 2 * kp]) = p;
    }
  }
}

// C = act(A(MxK,row) * B(KxN,row) + bias). ACT: 0 = none, 1 = relu
template <int ACT>
__global__ __launch_bounds__(256) void gemm_bias_act_wmma(
    const float* __restrict__ A, const float* __restrict__ B,
    const float* __restrict__ bias, float* __restrict__ C,
    int M, int N, int K) {
  __shared__ __bf16 AsBuf[2][BM * LDP];   // 2 x 10 KB
  __shared__ __bf16 BsBuf[2][BN * LDP];   // 2 x 20 KB

  const int tid  = threadIdx.x;
  const int lane = tid & 31;
  const int wave = tid >> 5;
  const int wm   = wave >> 2;   // 0..1 -> 64-row strip
  const int wn   = wave & 3;    // 0..3 -> 64-col strip
  const int bm0  = blockIdx.y * BM;
  const int bn0  = blockIdx.x * BN;

  const int fr   = lane & 15;   // row (A) / col (B,C) within 16
  const int hb   = lane >> 4;   // half-wave selector
  const int a_kb = hb * 16;     // A frag: K base 0 or 16
  const int b_kg = hb * 16;     // B frag: K group 0 or 16

  v8f_t acc[4][4];
#pragma unroll
  for (int i = 0; i < 4; ++i)
#pragma unroll
    for (int j = 0; j < 4; ++j)
      acc[i][j] = (v8f_t){0.f, 0.f, 0.f, 0.f, 0.f, 0.f, 0.f, 0.f};

  const int nk = K / BK;
  StageRegs r;

  // prologue: stage tile 0 into buffer 0
  stage_load_global(A, B, K, N, bm0, bn0, 0, tid, r);
  stage_store_lds(AsBuf[0], BsBuf[0], tid, r);
  __syncthreads();

  for (int kt = 0; kt < nk; ++kt) {
    const int cur = kt & 1;
    const bool has_next = (kt + 1) < nk;

    // 1) issue next tile's global loads (latency hidden under WMMA below)
    if (has_next)
      stage_load_global(A, B, K, N, bm0, bn0, (kt + 1) * BK, tid, r);

    // 2) compute current tile from LDS
    const __bf16* As = AsBuf[cur];
    const __bf16* Bs = BsBuf[cur];

    AFragU af[4];
#pragma unroll
    for (int fm = 0; fm < 4; ++fm) {
      int row = wm * 64 + fm * 16 + fr;
      af[fm].h[0] = *(const v8bf_t*)(&As[row * LDP + a_kb]);       // K kb..kb+7
      af[fm].h[1] = *(const v8bf_t*)(&As[row * LDP + a_kb + 8]);   // K kb+8..+15
    }
    AFragU bf[4];
#pragma unroll
    for (int fn = 0; fn < 4; ++fn) {
      int col = wn * 64 + fn * 16 + fr;
      bf[fn].h[0] = *(const v8bf_t*)(&Bs[col * LDP + b_kg]);       // K kg..kg+7
      bf[fn].h[1] = *(const v8bf_t*)(&Bs[col * LDP + b_kg + 8]);   // K kg+8..+15
    }
#pragma unroll
    for (int fm = 0; fm < 4; ++fm)
#pragma unroll
      for (int fn = 0; fn < 4; ++fn)
        acc[fm][fn] = __builtin_amdgcn_wmma_f32_16x16x32_bf16(
            false, af[fm].v, false, bf[fn].v, (short)0, acc[fm][fn],
            false, false);

    // 3) write next tile into the other LDS buffer
    if (has_next)
      stage_store_lds(AsBuf[1 - cur], BsBuf[1 - cur], tid, r);

    __syncthreads();
  }

  // epilogue: bias (+relu), f32 store
#pragma unroll
  for (int fn = 0; fn < 4; ++fn) {
    int n = bn0 + wn * 64 + fn * 16 + fr;
    float bv = bias[n];
#pragma unroll
    for (int fm = 0; fm < 4; ++fm) {
      int mbase = bm0 + wm * 64 + fm * 16 + hb * 8;
      float* cp = C + (size_t)mbase * N + n;
#pragma unroll
      for (int v = 0; v < 8; ++v) {
        float rr = acc[fm][fn][v] + bv;
        if (ACT == 1) rr = rr > 0.f ? rr : 0.f;
        cp[(size_t)v * N] = rr;
      }
    }
  }
}

// split z into mu/logvar and compute zs = mu + exp(0.5*logvar)*eps
__global__ __launch_bounds__(256) void vae_sample_kernel(
    const float* __restrict__ z, const float* __restrict__ eps,
    float* __restrict__ mu_out, float* __restrict__ lv_out,
    float* __restrict__ zs) {
  int i = blockIdx.x * 256 + threadIdx.x;   // over 4096*512
  int m = i >> 9;
  int n = i & 511;
  float mu = z[(size_t)m * 1024 + n];
  float lv = z[(size_t)m * 1024 + 512 + n];
  mu_out[i] = mu;
  lv_out[i] = lv;
  zs[i] = mu + expf(0.5f * lv) * eps[i];
}

extern "C" void kernel_launch(void* const* d_in, const int* in_sizes, int n_in,
                              void* d_out, int out_size, void* d_ws, size_t ws_size,
                              hipStream_t stream) {
  const float* x   = (const float*)d_in[0];
  const float* We1 = (const float*)d_in[1];
  const float* be1 = (const float*)d_in[2];
  const float* We2 = (const float*)d_in[3];
  const float* be2 = (const float*)d_in[4];
  const float* Wd1 = (const float*)d_in[5];
  const float* bd1 = (const float*)d_in[6];
  const float* Wd2 = (const float*)d_in[7];
  const float* bd2 = (const float*)d_in[8];
  const float* eps = (const float*)d_in[9];

  const size_t Bm = 4096, Din = 4096, H = 4096, LAT = 512;

  float* ws = (float*)d_ws;
  float* h  = ws;                        // 4096*4096
  float* z  = h  + Bm * H;               // 4096*1024
  float* zs = z  + Bm * 2 * LAT;         // 4096*512
  float* hd = zs + Bm * LAT;             // 4096*4096

  float* recon  = (float*)d_out;                 // 4096*4096
  float* mu_out = recon  + Bm * Din;             // 4096*512
  float* lv_out = mu_out + Bm * LAT;             // 4096*512

  dim3 blk(256);

  // h = relu(x @ W_e1 + b_e1)        [4096 x 4096 x 4096]
  dim3 g1(H / BN, Bm / BM);
  gemm_bias_act_wmma<1><<<g1, blk, 0, stream>>>(x, We1, be1, h, Bm, H, Din);

  // z = h @ W_e2 + b_e2              [4096 x 1024 x 4096]
  dim3 g2((2 * LAT) / BN, Bm / BM);
  gemm_bias_act_wmma<0><<<g2, blk, 0, stream>>>(h, We2, be2, z, Bm, 2 * LAT, H);

  // mu/logvar split + zs sampling
  vae_sample_kernel<<<(Bm * LAT) / 256, blk, 0, stream>>>(z, eps, mu_out, lv_out, zs);

  // hd = relu(zs @ W_d1 + b_d1)      [4096 x 4096 x 512]
  dim3 g3(H / BN, Bm / BM);
  gemm_bias_act_wmma<1><<<g3, blk, 0, stream>>>(zs, Wd1, bd1, hd, Bm, H, LAT);

  // recon = hd @ W_d2 + b_d2         [4096 x 4096 x 4096]
  dim3 g4(Din / BN, Bm / BM);
  gemm_bias_act_wmma<0><<<g4, blk, 0, stream>>>(hd, Wd2, bd2, recon, Bm, Din, H);
}
